// MultiQueryAttention_53772990546487
// MI455X (gfx1250) — matrix operand order repaired and checked by reading, not compiled, not run on hardware
//
#include <hip/hip_runtime.h>
#include <hip/hip_bf16.h>
#include <cmath>

// ---------------------------------------------------------------------------
// MultiQueryAttention (GQA, ALiBi, causal) for gfx1250 / MI455X.
// Matrix math: V_WMMA_F32_16X16X32_F16 (wave32 WMMA).
// GEMM1 tiles staged via GLOBAL_LOAD_ASYNC_TO_LDS_B128 (ASYNCcnt).
// GEMM2 tiles staged via the Tensor Data Mover: TENSOR_LOAD_TO_LDS (TENSORcnt).
// Both double-buffered, steady-state K-loop is branch-free (last step peeled).
// ---------------------------------------------------------------------------

typedef _Float16 half_t;
typedef __attribute__((ext_vector_type(8)))  _Float16 v8h;
typedef __attribute__((ext_vector_type(16))) _Float16 v16h;
typedef __attribute__((ext_vector_type(8)))  float    v8f;
typedef __attribute__((ext_vector_type(4)))  unsigned u32x4;
typedef __attribute__((ext_vector_type(8)))  unsigned u32x8;

#define S_LEN    2048
#define HID      2048
#define HEAD_DIM 128
#define NHEADS   16
#define NGROUPS  4
#define QKV_OUT  (HID + 2 * HEAD_DIM * NGROUPS)   // 3072

// ---- WMMA fragment loaders (contiguous k = 16*(lane/16) + s, same in A and B) ------

__device__ __forceinline__ v16h load_a_frag(const half_t* base,
                                            int lda, int row0, int k0, int lane) {
  int r   = row0 + (lane & 15);
  int sel = lane >> 4;
  const half_t* p = base + (long)r * lda + k0;
  v8h lo = *(const v8h*)(p + sel * 16);
  v8h hi = *(const v8h*)(p + sel * 16 + 8);
  return __builtin_shufflevector(lo, hi, 0,1,2,3,4,5,6,7,8,9,10,11,12,13,14,15);
}

__device__ __forceinline__ v16h load_b_frag(const half_t* base,
                                            int ldb, int col0, int k0, int lane) {
  int c   = col0 + (lane & 15);
  int sel = lane >> 4;
  return *(const v16h*)(base + (long)c * ldb + k0 + sel * 16);
}

__device__ __forceinline__ v8f wmma_f16(v16h a, v16h b, v8f c) {
  return __builtin_amdgcn_wmma_f32_16x16x32_f16(
      false, a, false, b, (short)0, c, false, false);
}

// ---- CDNA5 async copy to LDS (ASYNCcnt) ------------------------------------

__device__ __forceinline__ unsigned lds_off(const void* p) {
  // generic LDS pointer: addr[31:0] is the LDS byte offset (flat aperture rule)
  return (unsigned)(size_t)p;
}

__device__ __forceinline__ void async_ld_b128(unsigned lds_addr, const void* gaddr) {
  asm volatile("global_load_async_to_lds_b128 %0, %1, off"
               :: "v"(lds_addr), "v"(gaddr) : "memory");
}

__device__ __forceinline__ void wait_async_le(int keep4) {
  if (keep4) asm volatile("s_wait_asynccnt 0x4" ::: "memory");
  else       asm volatile("s_wait_asynccnt 0x0" ::: "memory");
}

// ---- CDNA5 Tensor Data Mover: 2D tile -> LDS (TENSORcnt) -------------------
// D# per CDNA5 ISA (2-group form, 2D tensor).

__device__ __forceinline__ void tdm_load_2d(unsigned lds_addr, const void* gaddr,
                                            unsigned tensor_d0, unsigned tensor_d1,
                                            unsigned tile_d0,  unsigned tile_d1,
                                            unsigned stride0_elems) {
  unsigned long ga = (unsigned long)gaddr;
  u32x4 g0;
  g0[0] = 1u;                                   // count=1, user mode, no gather
  g0[1] = lds_addr;                             // lds_addr [63:32]
  g0[2] = (unsigned)ga;                         // global_addr[31:0]
  g0[3] = (unsigned)(ga >> 32) | (2u << 30);    // global_addr[56:32] | type=2
  u32x8 g1;
  g1[0] = 1u << 16;                             // data_size=1 (2 bytes/elem)
  g1[1] = (tensor_d0 & 0xFFFFu) << 16;          // tensor_dim0[15:0]
  g1[2] = ((tensor_d0 >> 16) & 0xFFFFu) | ((tensor_d1 & 0xFFFFu) << 16);
  g1[3] = ((tensor_d1 >> 16) & 0xFFFFu) | (tile_d0 << 16);   // tile_dim0
  g1[4] = tile_d1;                              // tile_dim1 (tile_dim2=0)
  g1[5] = stride0_elems;                        // tensor_dim0_stride[31:0]
  g1[6] = 0;                                    // stride hi / dim1_stride
  g1[7] = 0;
  asm volatile("tensor_load_to_lds %0, %1" :: "s"(g0), "s"(g1) : "memory");
}

__device__ __forceinline__ void wait_tensor_le(int keep2) {
  if (keep2) __builtin_amdgcn_s_wait_tensorcnt((short)2);
  else       __builtin_amdgcn_s_wait_tensorcnt((short)0);
}

// ---- f32 -> f16 conversion -------------------------------------------------

__global__ void cvt_f32_f16(const float* __restrict__ src, half_t* __restrict__ dst, int n) {
  int i = blockIdx.x * blockDim.x + threadIdx.x;
  if (i < n) dst[i] = (half_t)src[i];
}

// ---- GEMM 1: xqkv = x @ wqkv^T, scatter into q / k / v^T -------------------
// Block 256 threads = 8 waves (4x2); block tile 128x128, BK=32; wave tile 32x64.
// Tiles double-buffered in LDS, filled by per-lane async-to-LDS copies.

__global__ __launch_bounds__(256)
void gemm_qkv(const half_t* __restrict__ A,   // [S][HID]  f16
              const half_t* __restrict__ B,   // [QKV_OUT][HID] f16
              half_t* __restrict__ qbuf,      // [NHEADS][S][HEAD_DIM]
              half_t* __restrict__ kbuf,      // [NGROUPS][S][HEAD_DIM]
              half_t* __restrict__ vtbuf) {   // [NGROUPS][HEAD_DIM][S]
  __shared__ __align__(16) half_t ldsA[2][128 * 32];
  __shared__ __align__(16) half_t ldsB[2][128 * 32];

  int tid  = threadIdx.x;
  int lane = tid & 31;
  int wave = tid >> 5;
  int wm = wave >> 1, wn = wave & 1;
  int m0 = blockIdx.x * 128;
  int n0 = blockIdx.y * 128;

  // One 128x32 f16 tile (8KB) per matrix = 512 b128 chunks; straight-line
  // issue of 2 chunks/thread/matrix -> 4 async ops in flight per wave/stage.
  auto issue = [&](int buf, int k0, int c) {
    int row = c >> 2;
    int kc  = (c & 3) * 8;
    async_ld_b128(lds_off(&ldsA[buf][0]) + c * 16,
                  A + (long)(m0 + row) * HID + k0 + kc);
    async_ld_b128(lds_off(&ldsB[buf][0]) + c * 16,
                  B + (long)(n0 + row) * HID + k0 + kc);
  };
  auto stage = [&](int buf, int k0) { issue(buf, k0, tid); issue(buf, k0, tid + 256); };

  v8f acc[2][4] = {};
  auto compute = [&](int buf) {
    const half_t* tA = ldsA[buf];
    const half_t* tB = ldsB[buf];
    v16h a0 = load_a_frag(tA, 32, wm * 32,      0, lane);
    v16h a1 = load_a_frag(tA, 32, wm * 32 + 16, 0, lane);
#pragma unroll
    for (int j = 0; j < 4; ++j) {
      v16h b = load_b_frag(tB, 32, wn * 64 + j * 16, 0, lane);
      acc[0][j] = wmma_f16(a0, b, acc[0][j]);
      acc[1][j] = wmma_f16(a1, b, acc[1][j]);
    }
  };

  const int KSTEPS = HID / 32;
  stage(0, 0);
  // steady state: branch-free body
  for (int s = 0; s < KSTEPS - 1; ++s) {
    int buf = s & 1;
    stage(buf ^ 1, (s + 1) * 32);
    wait_async_le(1);
    __syncthreads();
    compute(buf);
    __syncthreads();
  }
  // peeled final step
  wait_async_le(0);
  __syncthreads();
  compute((KSTEPS - 1) & 1);

  int half_sel = lane >> 4, nlane = lane & 15;
#pragma unroll
  for (int i = 0; i < 2; ++i)
#pragma unroll
    for (int j = 0; j < 4; ++j)
#pragma unroll
      for (int e = 0; e < 8; ++e) {
        int m = m0 + wm * 32 + i * 16 + e + 8 * half_sel;
        int n = n0 + wn * 64 + j * 16 + nlane;
        half_t val = (half_t)acc[i][j][e];
        if (n < HID) {                      // Q
          int h = n >> 7, d = n & 127;
          qbuf[((long)h * S_LEN + m) * HEAD_DIM + d] = val;
        } else {                            // K / V
          int r = n - HID;
          int g = r >> 8, c = r & 255;
          if (c < HEAD_DIM)
            kbuf[((long)g * S_LEN + m) * HEAD_DIM + c] = val;
          else                              // store V transposed: [g][d][s]
            vtbuf[((long)g * HEAD_DIM + (c - HEAD_DIM)) * S_LEN + m] = val;
        }
      }
}

// ---- Flash attention (ALiBi + causal), one wave per 16 query rows ----------

__global__ __launch_bounds__(128)
void attn_kernel(const half_t* __restrict__ q,   // [NHEADS][S][HEAD_DIM]
                 const half_t* __restrict__ k,   // [NGROUPS][S][HEAD_DIM]
                 const half_t* __restrict__ vt,  // [NGROUPS][HEAD_DIM][S]
                 half_t* __restrict__ attn) {    // [S][HID]
  int lane = threadIdx.x & 31;
  int wave = threadIdx.x >> 5;
  int h = blockIdx.y;
  int g = h >> 2;                         // q_per_kv = 4
  int q0 = blockIdx.x * 64 + wave * 16;
  float slope = exp2f(-0.5f * (float)(h + 1));   // ALiBi slope, 16 heads
  const float scale = 0.08838834764831845f;      // 1/sqrt(128)

  const half_t* qh = q  + (long)h * S_LEN * HEAD_DIM;
  const half_t* kg = k  + (long)g * S_LEN * HEAD_DIM;
  const half_t* vg = vt + (long)g * HEAD_DIM * S_LEN;

  int half_sel = lane >> 4, nlane = lane & 15;

  v16h qa[4];
#pragma unroll
  for (int kk = 0; kk < 4; ++kk) qa[kk] = load_a_frag(qh, HEAD_DIM, q0, kk * 32, lane);

  v8f o[8] = {};
  float mi[8], li[8];
#pragma unroll
  for (int e = 0; e < 8; ++e) { mi[e] = -INFINITY; li[e] = 0.f; }

  __shared__ __align__(32) half_t pbuf[4][16 * 32];
  half_t* myp = pbuf[wave];

  int kend = q0 + 16;                      // causal: keys [0, q0+15]
  for (int kb = 0; kb < kend; kb += 32) {
    // S = Q * K^T  (two 16x16 n-tiles, K-dim 128 = 4 chained steps)
    v8f s[2] = {};
#pragma unroll
    for (int t = 0; t < 2; ++t)
#pragma unroll
      for (int kk = 0; kk < 4; ++kk)
        s[t] = wmma_f16(qa[kk],
                        load_b_frag(kg, HEAD_DIM, kb + t * 16, kk * 32, lane),
                        s[t]);

    // scale + ALiBi bias + causal mask
#pragma unroll
    for (int t = 0; t < 2; ++t)
#pragma unroll
      for (int e = 0; e < 8; ++e) {
        int qi = q0 + e + 8 * half_sel;
        int ki = kb + t * 16 + nlane;
        int diff = qi - ki;
        float sv = s[t][e] * scale - slope * sqrtf((float)(diff > 0 ? diff : 0));
        s[t][e] = (diff < 0) ? -65504.0f : sv;   // FP16_MIN on masked entries
      }

    // row max across the 16 lanes holding each row (C layout: row = vgpr slot)
    float rm[8];
#pragma unroll
    for (int e = 0; e < 8; ++e) rm[e] = fmaxf(s[0][e], s[1][e]);
#pragma unroll
    for (int off = 1; off < 16; off <<= 1)
#pragma unroll
      for (int e = 0; e < 8; ++e)
        rm[e] = fmaxf(rm[e], __shfl_xor(rm[e], off, 32));

    float alpha[8];
#pragma unroll
    for (int e = 0; e < 8; ++e) {
      float mn = fmaxf(mi[e], rm[e]);
      alpha[e] = __expf(mi[e] - mn);
      mi[e] = mn;
    }
#pragma unroll
    for (int t = 0; t < 2; ++t)
#pragma unroll
      for (int e = 0; e < 8; ++e)
        s[t][e] = __expf(s[t][e] - mi[e]);

    float rs[8];
#pragma unroll
    for (int e = 0; e < 8; ++e) rs[e] = s[0][e] + s[1][e];
#pragma unroll
    for (int off = 1; off < 16; off <<= 1)
#pragma unroll
      for (int e = 0; e < 8; ++e)
        rs[e] += __shfl_xor(rs[e], off, 32);
#pragma unroll
    for (int e = 0; e < 8; ++e) li[e] = li[e] * alpha[e] + rs[e];
#pragma unroll
    for (int d = 0; d < 8; ++d)
#pragma unroll
      for (int e = 0; e < 8; ++e)
        o[d][e] *= alpha[e];

    // C-layout P tile -> LDS -> reload in A-fragment layout (wave-synchronous;
    // LDS ops are in-order within a wave, wave_barrier blocks compiler motion)
#pragma unroll
    for (int t = 0; t < 2; ++t)
#pragma unroll
      for (int e = 0; e < 8; ++e)
        myp[(e + 8 * half_sel) * 32 + t * 16 + nlane] = (half_t)s[t][e];
    __builtin_amdgcn_wave_barrier();
    asm volatile("" ::: "memory");
    v16h pa = load_a_frag(myp, 32, 0, 0, lane);
    __builtin_amdgcn_wave_barrier();

    // O += P * V   (V^T stored [d][s] -> contiguous B-fragments)
#pragma unroll
    for (int d = 0; d < 8; ++d)
      o[d] = wmma_f16(pa, load_b_frag(vg, S_LEN, d * 16, kb, lane), o[d]);
  }

  // normalize and store head output into [s][h*128+d]
#pragma unroll
  for (int d = 0; d < 8; ++d)
#pragma unroll
    for (int e = 0; e < 8; ++e) {
      int m = q0 + e + 8 * half_sel;
      attn[(long)m * HID + h * HEAD_DIM + d * 16 + nlane] = (half_t)(o[d][e] / li[e]);
    }
}

// ---- GEMM 2: out = attn @ wo^T (f32 output), TDM-staged tiles --------------

__global__ __launch_bounds__(256)
void gemm_out(const half_t* __restrict__ A,   // [S][HID] f16
              const half_t* __restrict__ B,   // [HID][HID] f16
              float* __restrict__ out) {      // [S][HID] f32
  __shared__ __align__(16) half_t ldsA[2][128 * 32];
  __shared__ __align__(16) half_t ldsB[2][128 * 32];

  int tid  = threadIdx.x;
  int lane = tid & 31;
  int wave = tid >> 5;
  int wm = wave >> 1, wn = wave & 1;
  int m0 = blockIdx.x * 128;
  int n0 = blockIdx.y * 128;
  bool leader = (wave == 0);     // TDM ignores EXEC -> gate per wave

  // One TDM 2D descriptor per tile: 128 rows x 32 halfwords, row stride HID.
  auto stage = [&](int buf, int k0) {
    tdm_load_2d(lds_off(&ldsA[buf][0]), A + (long)m0 * HID + k0,
                HID, S_LEN, 32, 128, HID);
    tdm_load_2d(lds_off(&ldsB[buf][0]), B + (long)n0 * HID + k0,
                HID, HID, 32, 128, HID);
  };

  v8f acc[2][4] = {};
  auto compute = [&](int buf) {
    const half_t* tA = ldsA[buf];
    const half_t* tB = ldsB[buf];
    v16h a0 = load_a_frag(tA, 32, wm * 32,      0, lane);
    v16h a1 = load_a_frag(tA, 32, wm * 32 + 16, 0, lane);
#pragma unroll
    for (int j = 0; j < 4; ++j) {
      v16h b = load_b_frag(tB, 32, wn * 64 + j * 16, 0, lane);
      acc[0][j] = wmma_f16(a0, b, acc[0][j]);
      acc[1][j] = wmma_f16(a1, b, acc[1][j]);
    }
  };

  const int KSTEPS = HID / 32;
  if (leader) stage(0, 0);
  // steady state: branch-free body
  for (int s = 0; s < KSTEPS - 1; ++s) {
    int buf = s & 1;
    if (leader) { stage(buf ^ 1, (s + 1) * 32); wait_tensor_le(1); }
    __syncthreads();
    compute(buf);
    __syncthreads();
  }
  // peeled final step
  if (leader) wait_tensor_le(0);
  __syncthreads();
  compute((KSTEPS - 1) & 1);

  int half_sel = lane >> 4, nlane = lane & 15;
#pragma unroll
  for (int i = 0; i < 2; ++i)
#pragma unroll
    for (int j = 0; j < 4; ++j)
#pragma unroll
      for (int e = 0; e < 8; ++e) {
        int m = m0 + wm * 32 + i * 16 + e + 8 * half_sel;
        int n = n0 + wn * 64 + j * 16 + nlane;
        out[(long)m * HID + n] = acc[i][j][e];
      }
}

// ---- launch ----------------------------------------------------------------

extern "C" void kernel_launch(void* const* d_in, const int* in_sizes, int n_in,
                              void* d_out, int out_size, void* d_ws, size_t ws_size,
                              hipStream_t stream) {
  const float* x    = (const float*)d_in[0];
  const float* wqkv = (const float*)d_in[1];
  const float* wo   = (const float*)d_in[2];
  float* out = (float*)d_out;

  char* ws = (char*)d_ws;
  size_t off = 0;
  auto take = [&](size_t nhalf) {
    half_t* p = (half_t*)(ws + off);
    off = (off + nhalf * sizeof(half_t) + 255) & ~(size_t)255;
    return p;
  };
  half_t* x_h    = take((size_t)S_LEN * HID);
  half_t* wqkv_h = take((size_t)QKV_OUT * HID);
  half_t* wo_h   = take((size_t)HID * HID);
  half_t* q_h    = take((size_t)NHEADS * S_LEN * HEAD_DIM);
  half_t* k_h    = take((size_t)NGROUPS * S_LEN * HEAD_DIM);
  half_t* vt_h   = take((size_t)NGROUPS * HEAD_DIM * S_LEN);
  half_t* attn_h = take((size_t)S_LEN * HID);

  int n1 = S_LEN * HID;
  int n2 = QKV_OUT * HID;
  cvt_f32_f16<<<(n1 + 255) / 256, 256, 0, stream>>>(x, x_h, n1);
  cvt_f32_f16<<<(n2 + 255) / 256, 256, 0, stream>>>(wqkv, wqkv_h, n2);
  cvt_f32_f16<<<(n1 + 255) / 256, 256, 0, stream>>>(wo, wo_h, n1);

  dim3 g1(S_LEN / 128, QKV_OUT / 128);         // 16 x 24
  gemm_qkv<<<g1, 256, 0, stream>>>(x_h, wqkv_h, q_h, k_h, vt_h);

  dim3 ga(S_LEN / 64, NHEADS);                 // 32 x 16
  attn_kernel<<<ga, 128, 0, stream>>>(q_h, k_h, vt_h, attn_h);

  dim3 g2(S_LEN / 128, HID / 128);             // 16 x 16
  gemm_out<<<g2, 256, 0, stream>>>(attn_h, wo_h, out);
}
